// Attention_11476152615401
// MI455X (gfx1250) — matrix operand order repaired and checked by reading, not compiled, not run on hardware
//
#include <hip/hip_runtime.h>
#include <hip/hip_bf16.h>

// ---------------------------------------------------------------------------
// Attention forward for gfx1250 (MI455X).  All contractions via
// v_wmma_f32_16x16x32_f16 (f32 -> f16 operands, f32 accumulation).
// Double-buffered LDS staging; f16 tile staging uses the CDNA5 async
// global->LDS DMA path (ASYNCcnt); V stored pre-transposed for the PV GEMM.
// ---------------------------------------------------------------------------

typedef _Float16 h16 __attribute__((ext_vector_type(16)));
typedef _Float16 h4  __attribute__((ext_vector_type(4)));
typedef _Float16 h2  __attribute__((ext_vector_type(2)));
typedef float    f8  __attribute__((ext_vector_type(8)));

union H16 { h16 v; h2 p[8]; _Float16 e[16]; };
union F8  { f8 v;  float e[8]; };

#define DIM      768
#define NH       12
#define HD       64
#define BATCH    8
#define SEQ      1024
#define MTOT     (BATCH * SEQ)          // 8192 rows
#define SCALE    0.125f                 // 64^-0.5

__device__ __forceinline__ f8 wmma16(h16 a, h16 b, f8 c) {
  return __builtin_amdgcn_wmma_f32_16x16x32_f16(
      /*neg_a=*/false, a, /*neg_b=*/false, b,
      /*c_mod=*/(short)0, c, /*reuse_a=*/false, /*reuse_b=*/false);
}

// Async 16-byte global -> LDS copy (GLOBAL_LOAD_ASYNC_TO_LDS_B128, ASYNCcnt).
// LDS destination = low 32 bits of the generic shared-memory address.
__device__ __forceinline__ void async_copy_b128(void* lds, const void* gptr) {
  asm volatile("global_load_async_to_lds_b128 %0, %1, off"
               :
               : "v"((unsigned int)(unsigned long long)lds),
                 "v"((unsigned long long)gptr)
               : "memory");
}
__device__ __forceinline__ void wait_async0() {
  asm volatile("s_wait_asynccnt 0" ::: "memory");
}

// A-fragment (16x32 f16, M x K): lane = M (mod 16); lane-half selects k-group.
// Source row-major [m][k], leading dim ld (halfs).  LDS or global.
__device__ __forceinline__ h16 load_a_frag(const _Float16* s, int ld, int m0, int k0) {
  const int lane = threadIdx.x & 31;
  const int m  = m0 + (lane & 15);
  const int kg = (lane >> 4) * 16;
  const _Float16* row = s + (size_t)m * ld + k0;
  H16 a;
#pragma unroll
  for (int v = 0; v < 4; ++v) a.p[v]     = *(const h2*)(row + kg + 2 * v);
#pragma unroll
  for (int v = 0; v < 4; ++v) a.p[4 + v] = *(const h2*)(row + kg + 8 + 2 * v);
  return a.v;
}

// B-fragment (32x16 f16, K x N): lane = N (mod 16); lane-half selects k-group.
// Source K-contiguous: element (k,n) at s[n*ld + k].
__device__ __forceinline__ h16 load_b_fragT(const _Float16* s, int ld, int k0, int n0) {
  const int lane = threadIdx.x & 31;
  const int n  = n0 + (lane & 15);
  const int kg = (lane >> 4) * 16;
  const _Float16* row = s + (size_t)n * ld + k0 + kg;
  H16 b;
#pragma unroll
  for (int v = 0; v < 8; ++v) b.p[v] = *(const h2*)(row + 2 * v);
  return b.v;
}

// ---------------------------------------------------------------------------
// Kernel 1: qkv = x @ w_qkv + b_qkv.
//   Q, K -> f16 [B][H][N][64];  V -> f16 TRANSPOSED [B][H][64][N].
// grid (2304/128, 8192/256), block 256 (8 waves, each 32 rows x 128 cols).
// (f32 -> f16 conversion happens in staging, so this path stays VALU-staged.)
// ---------------------------------------------------------------------------
__global__ __launch_bounds__(256) void qkv_gemm_kernel(
    const float* __restrict__ X, const float* __restrict__ W,
    const float* __restrict__ bias,
    _Float16* __restrict__ Qw, _Float16* __restrict__ Kw, _Float16* __restrict__ Vtw) {
  __shared__ _Float16 sA[2][256][40];   // [m][k] row-major
  __shared__ _Float16 sB[2][128][40];   // [n][k] transposed

  const int tid  = threadIdx.x;
  const int lane = tid & 31;
  const int wave = __builtin_amdgcn_readfirstlane(tid >> 5);   // scalar
  const int mBase = blockIdx.y * 256;
  const int nBase = blockIdx.x * 128;
  const int Ncols = 3 * DIM;

  auto stage = [&](int buf, int kt) {
#pragma unroll
    for (int p = 0; p < 8; ++p) {                 // A: 256x32 f32 -> f16
      const int r = p * 32 + (tid >> 3);
      const int c = (tid & 7) * 4;
      const float4 f = *(const float4*)(X + (size_t)(mBase + r) * DIM + kt + c);
      h4 hv = { (_Float16)f.x, (_Float16)f.y, (_Float16)f.z, (_Float16)f.w };
      *(h4*)&sA[buf][r][c] = hv;
    }
#pragma unroll
    for (int p = 0; p < 4; ++p) {                 // B: 32x128 f32 -> f16, transposed
      const int r = p * 8 + (tid >> 5);
      const int c = (tid & 31) * 4;
      const float4 f = *(const float4*)(W + (size_t)(kt + r) * Ncols + nBase + c);
      sB[buf][c + 0][r] = (_Float16)f.x;
      sB[buf][c + 1][r] = (_Float16)f.y;
      sB[buf][c + 2][r] = (_Float16)f.z;
      sB[buf][c + 3][r] = (_Float16)f.w;
    }
  };

  f8 acc[2][8];
#pragma unroll
  for (int i = 0; i < 2; ++i)
#pragma unroll
    for (int j = 0; j < 8; ++j) acc[i][j] = {};

  stage(0, 0);
  for (int kt = 0; kt < DIM; kt += 32) {
    const int buf = (kt >> 5) & 1;
    __syncthreads();
    if (kt + 32 < DIM) stage(buf ^ 1, kt + 32);
    const h16 a0 = load_a_frag(&sA[buf][0][0], 40, wave * 32, 0);
    const h16 a1 = load_a_frag(&sA[buf][0][0], 40, wave * 32 + 16, 0);
#pragma unroll
    for (int t = 0; t < 8; ++t) {
      const h16 b = load_b_fragT(&sB[buf][0][0], 40, 0, t * 16);
      acc[0][t] = wmma16(a0, b, acc[0][t]);
      acc[1][t] = wmma16(a1, b, acc[1][t]);
    }
  }

  // Epilogue.  All tile-base index math is wave-uniform -> SALU.
  const int bb = mBase >> 10;                 // 256-row tiles never cross batch
  const int r0 = (mBase & 1023) + wave * 32;  // scalar
#pragma unroll
  for (int sub = 0; sub < 2; ++sub) {
#pragma unroll
    for (int t = 0; t < 8; ++t) {
      F8 c; c.v = acc[sub][t];
      const int nb    = nBase + t * 16;       // uniform
      const int which = nb / DIM;             // scalar div
      const int rem   = nb % DIM;
      const int hh    = rem / HD;
      const int dd    = (rem % HD) + (lane & 15);
      const float bv  = bias[nb + (lane & 15)];
      const size_t headBase = ((size_t)bb * NH + hh) << 10;   // *SEQ
#pragma unroll
      for (int v = 0; v < 8; ++v) {
        const int nrow = r0 + sub * 16 + v + 8 * (lane >> 4);
        const _Float16 val = (_Float16)(c.e[v] + bv);
        if (which == 0)      Qw[(headBase + nrow) * HD + dd] = val;
        else if (which == 1) Kw[(headBase + nrow) * HD + dd] = val;
        else                 Vtw[headBase * HD + ((size_t)dd << 10) + nrow] = val;
      }
    }
  }
}

// ---------------------------------------------------------------------------
// Kernel 2: flash attention.  grid (16, 12, 8) = (qtile, head, batch),
// block 128 (4 waves, 16 query rows each).  K / V^T tiles staged by the
// async global->LDS DMA path, double buffered.
// ---------------------------------------------------------------------------
__global__ __launch_bounds__(128) void attn_kernel(
    const _Float16* __restrict__ Qw, const _Float16* __restrict__ Kw,
    const _Float16* __restrict__ Vtw, _Float16* __restrict__ Ow) {
  __shared__ _Float16 sK[2][32][72];       // [key][d]
  __shared__ _Float16 sVt[2][64][40];      // [d][key]
  __shared__ _Float16 sP[4][16][40];       // per-wave P tile [qrow][key]

  const int b  = blockIdx.z;
  const int h  = blockIdx.y;
  const int qt = blockIdx.x;
  const int tid  = threadIdx.x;
  const int lane = tid & 31;
  const int wave = __builtin_amdgcn_readfirstlane(tid >> 5);   // scalar

  const size_t headOff = (((size_t)b * NH + h) << 10) * HD;
  const _Float16* Qbh  = Qw  + headOff;
  const _Float16* Kbh  = Kw  + headOff;
  const _Float16* Vtbh = Vtw + headOff;    // [d][n]
  const int q0 = qt * 64 + wave * 16;

  auto stage = [&](int buf, int kb) {
    {
      const int r   = tid >> 2;
      const int seg = (tid & 3) * 16;
      const _Float16* g = Kbh + (size_t)(kb + r) * HD + seg;
      async_copy_b128(&sK[buf][r][seg],     g);
      async_copy_b128(&sK[buf][r][seg + 8], g + 8);
    }
    {
      const int d   = tid >> 1;
      const int seg = (tid & 1) * 16;
      const _Float16* g = Vtbh + ((size_t)d << 10) + kb + seg;
      async_copy_b128(&sVt[buf][d][seg],     g);
      async_copy_b128(&sVt[buf][d][seg + 8], g + 8);
    }
  };

  // Q fragments: 16 rows x 64 d -> two K=32 A-fragments (global, coalesces to b128)
  h16 aq[2];
  aq[0] = load_a_frag(Qbh + (size_t)q0 * HD, HD, 0, 0);
  aq[1] = load_a_frag(Qbh + (size_t)q0 * HD, HD, 0, 32);

  f8 oacc[4];
#pragma unroll
  for (int t = 0; t < 4; ++t) oacc[t] = {};
  float rmax[8], lsum[8];
#pragma unroll
  for (int v = 0; v < 8; ++v) { rmax[v] = -1e30f; lsum[v] = 0.0f; }

  stage(0, 0);
  for (int kb = 0; kb < SEQ; kb += 32) {
    const int buf = (kb >> 5) & 1;
    wait_async0();                 // our async stores for `buf` have landed
    __syncthreads();               // everyone's have
    if (kb + 32 < SEQ) stage(buf ^ 1, kb + 32);

    // S = Q @ K^T (16 q rows x 32 keys), f32
    F8 s0, s1;
    {
      f8 z0 = {}, z1 = {};
      z0 = wmma16(aq[0], load_b_fragT(&sK[buf][0][0], 72, 0, 0), z0);
      z0 = wmma16(aq[1], load_b_fragT(&sK[buf][0][0], 72, 32, 0), z0);
      z1 = wmma16(aq[0], load_b_fragT(&sK[buf][0][0], 72, 0, 16), z1);
      z1 = wmma16(aq[1], load_b_fragT(&sK[buf][0][0], 72, 32, 16), z1);
      s0.v = z0; s1.v = z1;
    }

    // Streaming softmax.  C layout: lane = key (mod 16), VGPR v = q row.
    float alpha[8];
#pragma unroll
    for (int v = 0; v < 8; ++v) {
      float a0 = s0.e[v] * SCALE;
      float a1 = s1.e[v] * SCALE;
      float mx = fmaxf(a0, a1);
#pragma unroll
      for (int m = 1; m < 16; m <<= 1) mx = fmaxf(mx, __shfl_xor(mx, m, 16));
      const float nm = fmaxf(rmax[v], mx);
      const float p0 = __expf(a0 - nm);
      const float p1 = __expf(a1 - nm);
      float rs = p0 + p1;
#pragma unroll
      for (int m = 1; m < 16; m <<= 1) rs += __shfl_xor(rs, m, 16);
      alpha[v] = __expf(rmax[v] - nm);
      lsum[v]  = lsum[v] * alpha[v] + rs;
      rmax[v]  = nm;
      s0.e[v]  = p0;
      s1.e[v]  = p1;
    }
#pragma unroll
    for (int t = 0; t < 4; ++t) {
      F8 o; o.v = oacc[t];
#pragma unroll
      for (int v = 0; v < 8; ++v) o.e[v] *= alpha[v];
      oacc[t] = o.v;
    }

    // Re-layout P (C layout -> A layout) through per-wave LDS tile.
    {
      const int hi = lane >> 4;
      const int nn = lane & 15;
#pragma unroll
      for (int v = 0; v < 8; ++v) {
        const int row = v + 8 * hi;
        sP[wave][row][nn]      = (_Float16)s0.e[v];
        sP[wave][row][16 + nn] = (_Float16)s1.e[v];
      }
      asm volatile("s_wait_dscnt 0" ::: "memory");
    }
    const h16 ap = load_a_frag(&sP[wave][0][0], 40, 0, 0);

    // O += P @ V  (V^T in LDS is exactly the B-fragment source layout)
#pragma unroll
    for (int t = 0; t < 4; ++t) {
      const h16 bv = load_b_fragT(&sVt[buf][0][0], 40, 0, t * 16);
      oacc[t] = wmma16(ap, bv, oacc[t]);
    }
  }

  float inv[8];
#pragma unroll
  for (int v = 0; v < 8; ++v) inv[v] = 1.0f / lsum[v];

  // Store f16 to [B][N][H*64]
#pragma unroll
  for (int t = 0; t < 4; ++t) {
    F8 o; o.v = oacc[t];
    const int d = t * 16 + (lane & 15);
#pragma unroll
    for (int v = 0; v < 8; ++v) {
      const int row = v + 8 * (lane >> 4);
      Ow[((size_t)(b * SEQ + q0 + row)) * DIM + h * HD + d] =
          (_Float16)(o.e[v] * inv[v]);
    }
  }
}

// ---------------------------------------------------------------------------
// Kernel 3: out = attn @ w_proj + b_proj (f32 out).
// grid (768/128, 8192/256), block 256 (8 waves, 32 rows x 128 cols each).
// A tile (f16) staged via async DMA; B tile converts f32 -> f16 in VALU.
// ---------------------------------------------------------------------------
__global__ __launch_bounds__(256) void proj_gemm_kernel(
    const _Float16* __restrict__ A, const float* __restrict__ W,
    const float* __restrict__ bias, float* __restrict__ Y) {
  __shared__ _Float16 sA[2][256][40];
  __shared__ _Float16 sB[2][128][40];

  const int tid  = threadIdx.x;
  const int lane = tid & 31;
  const int wave = __builtin_amdgcn_readfirstlane(tid >> 5);
  const int mBase = blockIdx.y * 256;
  const int nBase = blockIdx.x * 128;

  auto stage = [&](int buf, int kt) {
#pragma unroll
    for (int p = 0; p < 2; ++p) {                 // A (f16): async DMA, 256x32 halfs
      const int idx = p * 256 + tid;
      const int r   = idx >> 1;
      const int seg = (idx & 1) * 16;
      const _Float16* g = A + (size_t)(mBase + r) * DIM + kt + seg;
      async_copy_b128(&sA[buf][r][seg],     g);
      async_copy_b128(&sA[buf][r][seg + 8], g + 8);
    }
#pragma unroll
    for (int p = 0; p < 4; ++p) {                 // B: 32x128 f32 -> f16, transposed
      const int r = p * 8 + (tid >> 5);
      const int c = (tid & 31) * 4;
      const float4 f = *(const float4*)(W + (size_t)(kt + r) * DIM + nBase + c);
      sB[buf][c + 0][r] = (_Float16)f.x;
      sB[buf][c + 1][r] = (_Float16)f.y;
      sB[buf][c + 2][r] = (_Float16)f.z;
      sB[buf][c + 3][r] = (_Float16)f.w;
    }
  };

  f8 acc[2][8];
#pragma unroll
  for (int i = 0; i < 2; ++i)
#pragma unroll
    for (int j = 0; j < 8; ++j) acc[i][j] = {};

  stage(0, 0);
  for (int kt = 0; kt < DIM; kt += 32) {
    const int buf = (kt >> 5) & 1;
    wait_async0();
    __syncthreads();
    if (kt + 32 < DIM) stage(buf ^ 1, kt + 32);
    const h16 a0 = load_a_frag(&sA[buf][0][0], 40, wave * 32, 0);
    const h16 a1 = load_a_frag(&sA[buf][0][0], 40, wave * 32 + 16, 0);
#pragma unroll
    for (int t = 0; t < 8; ++t) {
      const h16 b = load_b_fragT(&sB[buf][0][0], 40, 0, t * 16);
      acc[0][t] = wmma16(a0, b, acc[0][t]);
      acc[1][t] = wmma16(a1, b, acc[1][t]);
    }
  }

#pragma unroll
  for (int sub = 0; sub < 2; ++sub) {
#pragma unroll
    for (int t = 0; t < 8; ++t) {
      F8 c; c.v = acc[sub][t];
      const int n = nBase + t * 16 + (lane & 15);
      const float bv = bias[n];
#pragma unroll
      for (int v = 0; v < 8; ++v) {
        const int m = mBase + wave * 32 + sub * 16 + v + 8 * (lane >> 4);
        Y[(size_t)m * DIM + n] = c.e[v] + bv;
      }
    }
  }
}

// ---------------------------------------------------------------------------
extern "C" void kernel_launch(void* const* d_in, const int* in_sizes, int n_in,
                              void* d_out, int out_size, void* d_ws, size_t ws_size,
                              hipStream_t stream) {
  const float* x      = (const float*)d_in[0];
  const float* w_qkv  = (const float*)d_in[1];
  const float* b_qkv  = (const float*)d_in[2];
  const float* w_proj = (const float*)d_in[3];
  const float* b_proj = (const float*)d_in[4];
  float* out = (float*)d_out;

  const size_t perTensor = (size_t)BATCH * NH * SEQ * HD;   // 6,291,456 halfs
  _Float16* Qw  = (_Float16*)d_ws;
  _Float16* Kw  = Qw + perTensor;
  _Float16* Vtw = Kw + perTensor;         // [B][H][64][N]
  _Float16* Aw  = Vtw + perTensor;        // [B][N][768]

  dim3 g1(3 * DIM / 128, MTOT / 256);     // (18, 32)
  qkv_gemm_kernel<<<g1, 256, 0, stream>>>(x, w_qkv, b_qkv, Qw, Kw, Vtw);

  dim3 g2(SEQ / 64, NH, BATCH);           // (16, 12, 8)
  attn_kernel<<<g2, 128, 0, stream>>>(Qw, Kw, Vtw, Aw);

  dim3 g3(DIM / 128, MTOT / 256);         // (6, 32)
  proj_gemm_kernel<<<g3, 256, 0, stream>>>(Aw, w_proj, b_proj, out);
}
